// DCNv2_63462436765991
// MI455X (gfx1250) — matrix operand order repaired and checked by reading, not compiled, not run on hardware
//
#include <hip/hip_runtime.h>
#include <hip/hip_bf16.h>

typedef __attribute__((ext_vector_type(16))) __bf16 v16bf;
typedef __attribute__((ext_vector_type(8)))  float  v8f;
typedef int v4i __attribute__((vector_size(16)));

#define BATCH   16384
#define EMB     64
#define NNUM    13
#define NCAT    26
#define CVOCAB  10000
#define DDIM    2624      // (2 + 13 + 26) * 64
#define MLPD    1024
#define LCROSS  4

// LDS tile geometry (units: bf16 halves). Row stride padded 32 -> 40 halves
// (80 B) to spread DS bank groups across lanes.
#define LDS_STRIDE 40
#define A_ROWS     256
#define B_COLS     64
#define B_OFF      (A_ROWS * LDS_STRIDE)              // 10240 halves
#define BUF_HALVES (B_OFF + B_COLS * LDS_STRIDE)      // 12800 halves (25 KB)

union FragU { uint4 q[2]; v16bf v; };

#if defined(__gfx1250__) && __has_builtin(__builtin_amdgcn_global_load_async_to_lds_b128)
#define HAVE_ASYNC_LDS 1
#endif

__device__ __forceinline__ void cp16_async(const __bf16* g, __bf16* l) {
#ifdef HAVE_ASYNC_LDS
    __builtin_amdgcn_global_load_async_to_lds_b128(
        (__attribute__((address_space(1))) v4i*)(g),
        (__attribute__((address_space(3))) v4i*)(l),
        0, 0);
#else
    (void)g; (void)l;
#endif
}

__device__ __forceinline__ void wait_async_zero() {
#if defined(__gfx1250__) && __has_builtin(__builtin_amdgcn_s_wait_asynccnt)
    __builtin_amdgcn_s_wait_asynccnt(0);
#elif defined(__gfx1250__)
    asm volatile("s_wait_asynccnt 0x0" ::: "memory");
#endif
}

// A fragment (16x32 bf16 MxK) from LDS: lane l (m=l&15, g=l>>4):
//   halves 0..7 -> k = g*8 + (0..7), halves 8..15 -> k = 16 + g*8 + (0..7)
__device__ __forceinline__ v16bf fragA_lds(const __bf16* rowp, int g) {
    const __bf16* p = rowp + g * 8;
    FragU f;
    f.q[0] = *(const uint4*)(p);
    f.q[1] = *(const uint4*)(p + 16);
    return f.v;
}

// B fragment (32x16 bf16 KxN) from LDS (stored as W^T columns):
// lane l (n=l&15, g=l>>4): halves 0..7 -> k = g*8 + (0..7), halves 8..15 -> k = 16 + g*8 + (0..7)
__device__ __forceinline__ v16bf fragB_lds(const __bf16* colp, int g) {
    const __bf16* p = colp + g * 8;
    FragU f;
    f.q[0] = *(const uint4*)(p);
    f.q[1] = *(const uint4*)(p + 16);
    return f.v;
}

// MODE 0: cross epilogue  xf[m,n] = x0[m,n]*(acc+bias[n]) + xf[m,n]; outb = bf16(xf)
// MODE 1: relu epilogue   outb[m,n] = bf16(max(acc+bias[n], 0))
// Workgroup: 256 threads = 8 waves, arranged 8(M) x 1(N).
// WG tile 256(M) x 64(N); wave tile 32(M) x 64(N) = 8 WMMA accumulators.
template <int MODE>
__global__ __launch_bounds__(256) void gemm_wmma_kernel(
    const __bf16* __restrict__ A,    // [M,K] bf16 row-major
    const __bf16* __restrict__ WT,   // [N,K] bf16 row-major (W transposed)
    const float*  __restrict__ bias, // [N]
    const __bf16* __restrict__ x0b,  // [M,N] bf16 (cross only)
    float*        __restrict__ xf,   // [M,N] f32 in/out (cross only)
    __bf16*       __restrict__ outb, // [M,N] bf16 out
    int M, int N, int K)
{
    __shared__ __align__(32) __bf16 smem[2 * BUF_HALVES];

    const int tid    = threadIdx.x;
    const int lane   = tid & 31;
    const int wave   = tid >> 5;
    const int mBlock = blockIdx.x * 256;
    const int nBlock = blockIdx.y * 64;

    // ---- staging assignment: 16B chunks ----
    // A tile: 256 rows x 32 halves = 1024 chunks; thread handles rows rA + {0,64,128,192}.
    // B tile:  64 cols x 32 halves =  256 chunks; thread handles one.
    const int rA = tid >> 2, oA = (tid & 3) * 8;
    const int cB = tid >> 2, oB = (tid & 3) * 8;

    const __bf16* gA = A  + (size_t)(mBlock + rA) * K + oA;
    const __bf16* gB = WT + (size_t)(nBlock + cB) * K + oB;
    const size_t  gAstep = (size_t)64 * K;        // 64 rows in halves

    const int lA = rA * LDS_STRIDE + oA;          // + p * 64*LDS_STRIDE
    const int lB = B_OFF + cB * LDS_STRIDE + oB;

    const int g    = lane >> 4;
    const int l15  = lane & 15;
    const int nK   = K / 32;

    v8f acc0[4] = {};
    v8f acc1[4] = {};

    const int aOff0 = (wave * 32 + l15) * LDS_STRIDE;
    const int aOff1 = (wave * 32 + 16 + l15) * LDS_STRIDE;

#ifdef HAVE_ASYNC_LDS
    // prologue: fill buffer 0
#pragma unroll
    for (int p = 0; p < 4; ++p)
        cp16_async(gA + p * gAstep, smem + lA + p * 64 * LDS_STRIDE);
    cp16_async(gB, smem + lB);

    int buf = 0;
    for (int kt = 0; kt < nK - 1; ++kt) {          // steady state: branch-free body
        wait_async_zero();
        __syncthreads();
        {
            const int kb = (kt + 1) * 32;
            __bf16* nb = smem + (buf ^ 1) * BUF_HALVES;
#pragma unroll
            for (int p = 0; p < 4; ++p)
                cp16_async(gA + p * gAstep + kb, nb + lA + p * 64 * LDS_STRIDE);
            cp16_async(gB + kb, nb + lB);
        }
        const __bf16* sb = smem + buf * BUF_HALVES;
        const v16bf a0 = fragA_lds(sb + aOff0, g);
        const v16bf a1 = fragA_lds(sb + aOff1, g);
#pragma unroll
        for (int j = 0; j < 4; ++j) {
            const v16bf b = fragB_lds(sb + B_OFF + (j * 16 + l15) * LDS_STRIDE, g);
            acc0[j] = __builtin_amdgcn_wmma_f32_16x16x32_bf16(
                          false, a0, false, b, (short)0, acc0[j], false, false);
            acc1[j] = __builtin_amdgcn_wmma_f32_16x16x32_bf16(
                          false, a1, false, b, (short)0, acc1[j], false, false);
        }
        __syncthreads();   // all reads of `buf` done before it is refilled
        buf ^= 1;
    }
    // epilogue K-step (no further prefetch)
    {
        wait_async_zero();
        __syncthreads();
        const __bf16* sb = smem + buf * BUF_HALVES;
        const v16bf a0 = fragA_lds(sb + aOff0, g);
        const v16bf a1 = fragA_lds(sb + aOff1, g);
#pragma unroll
        for (int j = 0; j < 4; ++j) {
            const v16bf b = fragB_lds(sb + B_OFF + (j * 16 + l15) * LDS_STRIDE, g);
            acc0[j] = __builtin_amdgcn_wmma_f32_16x16x32_bf16(
                          false, a0, false, b, (short)0, acc0[j], false, false);
            acc1[j] = __builtin_amdgcn_wmma_f32_16x16x32_bf16(
                          false, a1, false, b, (short)0, acc1[j], false, false);
        }
    }
#else
    // fallback: synchronous VGPR -> LDS staging, single buffer
    for (int kt = 0; kt < nK; ++kt) {
        const int kb = kt * 32;
        uint4 tA[4];
#pragma unroll
        for (int p = 0; p < 4; ++p) tA[p] = *(const uint4*)(gA + p * gAstep + kb);
        const uint4 tB = *(const uint4*)(gB + kb);
        __syncthreads();
#pragma unroll
        for (int p = 0; p < 4; ++p) *(uint4*)(smem + lA + p * 64 * LDS_STRIDE) = tA[p];
        *(uint4*)(smem + lB) = tB;
        __syncthreads();
        const v16bf a0 = fragA_lds(smem + aOff0, g);
        const v16bf a1 = fragA_lds(smem + aOff1, g);
#pragma unroll
        for (int j = 0; j < 4; ++j) {
            const v16bf b = fragB_lds(smem + B_OFF + (j * 16 + l15) * LDS_STRIDE, g);
            acc0[j] = __builtin_amdgcn_wmma_f32_16x16x32_bf16(
                          false, a0, false, b, (short)0, acc0[j], false, false);
            acc1[j] = __builtin_amdgcn_wmma_f32_16x16x32_bf16(
                          false, a1, false, b, (short)0, acc1[j], false, false);
        }
    }
#endif

    // C/D layout: element r -> m = tileM + r + 8*(lane>>4), n = tileN + (lane&15)
    const int mB0 = mBlock + wave * 32 + g * 8;
#pragma unroll
    for (int j = 0; j < 4; ++j) {
        const int n  = nBlock + j * 16 + l15;
        const float bn = bias[n];
#pragma unroll
        for (int r = 0; r < 8; ++r) {
#pragma unroll
            for (int half = 0; half < 2; ++half) {
                const int m = mB0 + half * 16 + r;
                const size_t off = (size_t)m * N + n;
                const float av = (half == 0) ? acc0[j][r] : acc1[j][r];
                const float t = av + bn;
                if (MODE == 0) {
                    const float res = (float)x0b[off] * t + xf[off];
                    xf[off]   = res;          // in place: unique owner per element
                    outb[off] = (__bf16)res;  // next layer's GEMM input
                } else {
                    const float res = t > 0.0f ? t : 0.0f;
                    outb[off] = (__bf16)res;
                }
            }
        }
    }
}

__global__ void embed_kernel(const int* __restrict__ user_in, const int* __restrict__ item_in,
                             const float* __restrict__ numf, const int* __restrict__ catf,
                             const float* __restrict__ uemb, const float* __restrict__ iemb,
                             const float* __restrict__ ctab,
                             const float* __restrict__ numW, const float* __restrict__ numB,
                             float* __restrict__ xf, __bf16* __restrict__ x0b)
{
    const long idx = (long)blockIdx.x * blockDim.x + threadIdx.x;
    if (idx >= (long)BATCH * DDIM) return;
    const int b   = (int)(idx / DDIM);
    const int d   = (int)(idx % DDIM);
    const int seg = d >> 6;
    const int e   = d & 63;
    float v;
    if (seg == 0) {
        v = uemb[(size_t)user_in[b] * EMB + e];
    } else if (seg == 1) {
        v = iemb[(size_t)item_in[b] * EMB + e];
    } else if (seg < 2 + NNUM) {
        const int f = seg - 2;
        v = numf[b * NNUM + f] * numW[f * EMB + e] + numB[f * EMB + e];
    } else {
        const int f = seg - 2 - NNUM;
        const int c = catf[b * NCAT + f];
        v = ctab[((size_t)f * CVOCAB + c) * EMB + e];
    }
    xf[idx]  = v;           // evolving cross state (starts at x0)
    x0b[idx] = (__bf16)v;   // constant x0 in bf16 (GEMM input + epilogue term)
}

// WT[layer][n][k] = (bf16) W[layer][k][n]
__global__ void transpose_bf16_kernel(const float* __restrict__ W, __bf16* __restrict__ WT,
                                      int K, int N, long total)
{
    const long idx = (long)blockIdx.x * blockDim.x + threadIdx.x;
    if (idx >= total) return;
    const long per   = (long)K * N;
    const long layer = idx / per;
    const long r     = idx % per;
    const int  n     = (int)(r / K);
    const int  k     = (int)(r % K);
    WT[idx] = (__bf16)W[(size_t)layer * per + (size_t)k * N + n];
}

__global__ __launch_bounds__(256) void final_kernel(
    const float* __restrict__ xf, const __bf16* __restrict__ hb,
    const float* __restrict__ Wf, const float* __restrict__ bfin,
    float* __restrict__ out)
{
    const int wv   = (int)((blockIdx.x * 256 + threadIdx.x) >> 5);
    const int lane = threadIdx.x & 31;
    if (wv >= BATCH) return;
    float s = 0.0f;
    const float* xr = xf + (size_t)wv * DDIM;
    for (int d = lane; d < DDIM; d += 32) s += xr[d] * Wf[d];
    const __bf16* hr = hb + (size_t)wv * MLPD;
    for (int j = lane; j < MLPD; j += 32) s += (float)hr[j] * Wf[DDIM + j];
#pragma unroll
    for (int o = 16; o; o >>= 1) s += __shfl_xor(s, o, 32);
    if (lane == 0) {
        const float z = s + bfin[0];
        out[wv] = 1.0f / (1.0f + __expf(-z));
    }
}

extern "C" void kernel_launch(void* const* d_in, const int* in_sizes, int n_in,
                              void* d_out, int out_size, void* d_ws, size_t ws_size,
                              hipStream_t stream) {
    const int*   user_in = (const int*)  d_in[0];
    const int*   item_in = (const int*)  d_in[1];
    const float* numf    = (const float*)d_in[2];
    const int*   catf    = (const int*)  d_in[3];
    const float* uemb    = (const float*)d_in[4];
    const float* iemb    = (const float*)d_in[5];
    const float* ctab    = (const float*)d_in[6];
    const float* numW    = (const float*)d_in[7];
    const float* numB    = (const float*)d_in[8];
    const float* Wc      = (const float*)d_in[9];
    const float* bc      = (const float*)d_in[10];
    const float* W0      = (const float*)d_in[11];
    const float* b0      = (const float*)d_in[12];
    const float* Wh      = (const float*)d_in[13];
    const float* bh      = (const float*)d_in[14];
    const float* Wf      = (const float*)d_in[15];
    const float* bfin    = (const float*)d_in[16];
    (void)in_sizes; (void)n_in; (void)out_size; (void)ws_size;

    char* ws = (char*)d_ws;
    size_t off = 0;
    auto carve = [&](size_t bytes) -> char* {
        char* p = ws + off;
        off += (bytes + 255) & ~(size_t)255;
        return p;
    };
    float*  xf  = (float*) carve((size_t)BATCH * DDIM * 4);
    __bf16* x0b = (__bf16*)carve((size_t)BATCH * DDIM * 2);
    __bf16* xb0 = (__bf16*)carve((size_t)BATCH * DDIM * 2);
    __bf16* xb1 = (__bf16*)carve((size_t)BATCH * DDIM * 2);
    __bf16* hb0 = (__bf16*)carve((size_t)BATCH * MLPD * 2);
    __bf16* hb1 = (__bf16*)carve((size_t)BATCH * MLPD * 2);
    __bf16* WcT = (__bf16*)carve((size_t)LCROSS * DDIM * DDIM * 2);
    __bf16* W0T = (__bf16*)carve((size_t)DDIM * MLPD * 2);
    __bf16* WhT = (__bf16*)carve((size_t)3 * MLPD * MLPD * 2);

    // Weight conversion + transpose to bf16 [N,K]
    {
        long tot = (long)LCROSS * DDIM * DDIM;
        transpose_bf16_kernel<<<(int)((tot + 255) / 256), 256, 0, stream>>>(Wc, WcT, DDIM, DDIM, tot);
        tot = (long)MLPD * DDIM;
        transpose_bf16_kernel<<<(int)((tot + 255) / 256), 256, 0, stream>>>(W0, W0T, DDIM, MLPD, tot);
        tot = 3L * MLPD * MLPD;
        transpose_bf16_kernel<<<(int)((tot + 255) / 256), 256, 0, stream>>>(Wh, WhT, MLPD, MLPD, tot);
    }
    // Embedding gather + numeric linears -> x0 (f32 state + bf16 GEMM input)
    {
        long tot = (long)BATCH * DDIM;
        embed_kernel<<<(int)((tot + 255) / 256), 256, 0, stream>>>(
            user_in, item_in, numf, catf, uemb, iemb, ctab, numW, numB, xf, x0b);
    }
    // Cross network: 4 fused GEMM+epilogue layers (bf16 ping-pong, f32 in place)
    {
        dim3 grid(BATCH / 256, DDIM / 64);
        const __bf16* xin = x0b;
        __bf16* outs[LCROSS] = {xb0, xb1, xb0, xb1};
        for (int i = 0; i < LCROSS; ++i) {
            gemm_wmma_kernel<0><<<grid, 256, 0, stream>>>(
                xin, WcT + (size_t)i * DDIM * DDIM, bc + (size_t)i * DDIM,
                x0b, xf, outs[i], BATCH, DDIM, DDIM);
            xin = outs[i];
        }
    }
    // Deep MLP: D->1024 then 3x 1024->1024, ReLU fused
    const __bf16* hin;
    {
        dim3 grid(BATCH / 256, MLPD / 64);
        gemm_wmma_kernel<1><<<grid, 256, 0, stream>>>(
            x0b, W0T, b0, nullptr, nullptr, hb0, BATCH, MLPD, DDIM);
        hin = hb0;
        __bf16* houts[3] = {hb1, hb0, hb1};
        for (int i = 0; i < 3; ++i) {
            gemm_wmma_kernel<1><<<grid, 256, 0, stream>>>(
                hin, WhT + (size_t)i * MLPD * MLPD, bh + (size_t)i * MLPD,
                nullptr, nullptr, houts[i], BATCH, MLPD, MLPD);
            hin = houts[i];
        }
    }
    // Final concat-dot + sigmoid: one wave per batch row
    final_kernel<<<BATCH / 8, 256, 0, stream>>>(xf, hin, Wf, bfin, (float*)d_out);
}
